// Model_57655640981686
// MI455X (gfx1250) — matrix operand order repaired and checked by reading, not compiled, not run on hardware
//
#include <hip/hip_runtime.h>
#include <hip/hip_bf16.h>

// ---------------------------------------------------------------------------
// LAS (pyramidal BiLSTM encoder + attention decoder) for MI455X / gfx1250.
// bf16 WMMA (f32 accum) for all GEMM work; persistent LDS-resident kernels
// for the sequential recurrences (320KB LDS per WGP makes the 32x1200 f32
// gate buffer + an async-staged preGate row fit on-chip).
// ---------------------------------------------------------------------------

typedef __bf16 bhalf;
typedef __attribute__((ext_vector_type(16))) __bf16 v16bf;
typedef __attribute__((ext_vector_type(8)))  float  v8f;
typedef __attribute__((__vector_size__(4 * sizeof(int)))) int v4i;

#define DEV __device__ __forceinline__

// CDNA5 async global->LDS copy path (ASYNCcnt-tracked), with fallback.
#if defined(__has_builtin)
#if __has_builtin(__builtin_amdgcn_global_load_async_to_lds_b128) && \
    __has_builtin(__builtin_amdgcn_s_wait_asynccnt)
#define USE_ASYNC_LDS 1
#endif
#endif
#ifndef USE_ASYNC_LDS
#define USE_ASYNC_LDS 0
#endif

DEV float sigmoidf_(float x) { return 1.f / (1.f + __expf(-x)); }

DEV v8f wmma_bf16(v16bf a, v16bf b, v8f c) {
  return __builtin_amdgcn_wmma_f32_16x16x32_bf16(false, a, false, b, (short)0, c,
                                                 false, false);
}

#if USE_ASYNC_LDS
DEV void async_b128(const bhalf* gsrc, bhalf* ldst) {
  __builtin_amdgcn_global_load_async_to_lds_b128(
      (v4i*)const_cast<bhalf*>(gsrc), (v4i*)ldst, 0, 0);
}
#endif

// A-matrix fragment (16x32 bf16): lane holds row m=lane&15;
// K = 2v + 16*(lane>=16) (matches B fragment order).
DEV v16bf load_fragA(const bhalf* base, int ld) {
  int lane = threadIdx.x & 31;
  const bhalf* row = base + (size_t)(lane & 15) * ld;
  int koff = (lane & 16) ? 16 : 0;
  v16bf f;
#pragma unroll
  for (int v = 0; v < 8; ++v) {
    int k = (v << 1) + koff;
    f[2 * v]     = row[k];
    f[2 * v + 1] = row[k + 1];
  }
  return f;
}

// B-matrix fragment (32x16 bf16): lane holds col n=lane&15; K = 2v + 16*(lane>=16).
// Source is row-major W[N,Kp]; B[k][n] = W[n][k].
DEV v16bf load_fragB(const bhalf* base, int ld) {
  int lane = threadIdx.x & 31;
  const bhalf* row = base + (size_t)(lane & 15) * ld;
  int koff = (lane & 16) ? 16 : 0;
  v16bf f;
#pragma unroll
  for (int v = 0; v < 8; ++v) {
    int k = (v << 1) + koff;
    f[2 * v]     = row[k];
    f[2 * v + 1] = row[k + 1];
  }
  return f;
}

DEV v16bf load_fragB_cl(const bhalf* W, int n0, int N, int ld, int kb) {
  int lane = threadIdx.x & 31;
  int n = n0 + (lane & 15);
  if (n >= N) n = N - 1;  // values never stored; clamp keeps loads in-bounds
  const bhalf* row = W + (size_t)n * ld + kb;
  int koff = (lane & 16) ? 16 : 0;
  v16bf f;
#pragma unroll
  for (int v = 0; v < 8; ++v) {
    int k = (v << 1) + koff;
    f[2 * v]     = row[k];
    f[2 * v + 1] = row[k + 1];
  }
  return f;
}

// M=32 GEMM helper: C[32,N] = A[32,Kp] * W[N,Kp]^T (+bias) (+addB), tiles
// distributed across all waves of the block. A / addB may live in LDS.
DEV void mm32(const bhalf* A, int lda, const bhalf* W, int ldw,
              const float* bias, int biasN,
              const bhalf* addB, int ldadd,
              float* C, int ldc, int nTiles, int kSteps) {
  int nw = blockDim.x >> 5;
  int wid = threadIdx.x >> 5;
  int lane = threadIdx.x & 31;
  int nL = lane & 15, mOff = (lane & 16) ? 8 : 0;
  for (int tile = wid; tile < 2 * nTiles; tile += nw) {
    int mt = tile & 1, nt = tile >> 1;
    v8f acc;
#pragma unroll
    for (int i = 0; i < 8; ++i) acc[i] = 0.f;
    const bhalf* Ab = A + (size_t)(mt * 16) * lda;
    const bhalf* Wb = W + (size_t)(nt * 16) * ldw;
    for (int kk = 0; kk < kSteps; ++kk) {
      if (kk + 1 < kSteps) __builtin_prefetch(Wb + (kk + 1) * 32, 0, 3);
      v16bf fa = load_fragA(Ab + kk * 32, lda);
      v16bf fb = load_fragB(Wb + kk * 32, ldw);
      acc = wmma_bf16(fa, fb, acc);
    }
    int n = nt * 16 + nL;
    float bv = 0.f;
    if (bias) bv = (n < biasN) ? bias[n] : 0.f;
#pragma unroll
    for (int v = 0; v < 8; ++v) {
      int m = mt * 16 + v + mOff;
      float r = acc[v] + bv;
      if (addB) r += (float)addB[(size_t)m * ldadd + n];
      C[m * ldc + n] = r;
    }
  }
}

// ---------------------------------------------------------------------------
// Phase A: big throughput GEMM. C[M,N] = A[M,Kp] * W[N,Kp]^T + bias1 + bias2.
// Block = 8 waves -> 128x128 tile; wave -> 32x64 (8 accumulators).
// M must be a multiple of 128 (all call sites satisfy this).
// ---------------------------------------------------------------------------
__global__ __launch_bounds__(256) void wmma_gemm_kernel(
    const bhalf* __restrict__ A, const bhalf* __restrict__ W,
    const float* __restrict__ bias1, const float* __restrict__ bias2,
    float* __restrict__ Cf, bhalf* __restrict__ Cb, int N, int Kp) {
  int wid = threadIdx.x >> 5, lane = threadIdx.x & 31;
  int mBase = blockIdx.y * 128 + (wid & 3) * 32;
  int nBase = blockIdx.x * 128 + (wid >> 2) * 64;
  v8f acc[2][4];
#pragma unroll
  for (int i = 0; i < 2; ++i)
#pragma unroll
    for (int j = 0; j < 4; ++j)
#pragma unroll
      for (int v = 0; v < 8; ++v) acc[i][j][v] = 0.f;
  int kSteps = Kp >> 5;
  const bhalf* Abase = A + (size_t)mBase * Kp;
  for (int kk = 0; kk < kSteps; ++kk) {
    if (kk + 1 < kSteps) __builtin_prefetch(Abase + (kk + 1) * 32, 0, 3);
    v16bf fa0 = load_fragA(Abase + kk * 32, Kp);
    v16bf fa1 = load_fragA(Abase + (size_t)16 * Kp + kk * 32, Kp);
#pragma unroll
    for (int j = 0; j < 4; ++j) {
      v16bf fb = load_fragB_cl(W, nBase + j * 16, N, Kp, kk * 32);
      acc[0][j] = wmma_bf16(fa0, fb, acc[0][j]);
      acc[1][j] = wmma_bf16(fa1, fb, acc[1][j]);
    }
  }
  int nL = lane & 15, mOff = (lane & 16) ? 8 : 0;
#pragma unroll
  for (int j = 0; j < 4; ++j) {
    int n = nBase + j * 16 + nL;
    if (n >= N) continue;
    float bv = (bias1 ? bias1[n] : 0.f) + (bias2 ? bias2[n] : 0.f);
#pragma unroll
    for (int i = 0; i < 2; ++i)
#pragma unroll
      for (int v = 0; v < 8; ++v) {
        int m = mBase + i * 16 + v + mOff;
        float r = acc[i][j][v] + bv;
        size_t off = (size_t)m * N + n;
        if (Cf) Cf[off] = r;
        else    Cb[off] = (bhalf)r;
      }
  }
}

// ---------------------------------------------------------------------------
// Packing kernels (f32 -> K-padded bf16 layouts)
// ---------------------------------------------------------------------------
__global__ void pack_f32_kernel(const float* __restrict__ src, bhalf* __restrict__ dst,
                                int K, int Kp, size_t total) {
  size_t i = (size_t)blockIdx.x * blockDim.x + threadIdx.x;
  if (i >= total) return;
  size_t n = i / Kp;
  int k = (int)(i - n * Kp);
  dst[i] = (k < K) ? (bhalf)src[n * K + k] : (bhalf)0.f;
}

__global__ void pack_bf16_kernel(const bhalf* __restrict__ src, bhalf* __restrict__ dst,
                                 int K, int Kp, size_t total) {
  size_t i = (size_t)blockIdx.x * blockDim.x + threadIdx.x;
  if (i >= total) return;
  size_t n = i / Kp;
  int k = (int)(i - n * Kp);
  dst[i] = (k < K) ? src[n * K + k] : (bhalf)0.f;
}

// hseq[T,32,600] -> A[(T/2)*32, 1216]: row (t2,b) = [h[2t2,b] ; h[2t2+1,b] ; 0pad]
__global__ void pyramid_pack_kernel(const bhalf* __restrict__ src,
                                    bhalf* __restrict__ dst, int Thalf) {
  size_t total = (size_t)Thalf * 32 * 1216;
  size_t i = (size_t)blockIdx.x * blockDim.x + threadIdx.x;
  if (i >= total) return;
  size_t r = i / 1216;
  int k = (int)(i - r * 1216);
  int t2 = (int)(r >> 5), b = (int)(r & 31);
  bhalf v = (bhalf)0.f;
  if (k < 1200) {
    int half = (k >= 600) ? 1 : 0;
    int kk = k - half * 600;
    v = src[(((size_t)(2 * t2 + half)) * 32 + b) * 600 + kk];
  }
  dst[i] = v;
}

// Decoder cell weight: [1200,864]: cols 0..399 Wih[:,:400] (emb), 400..699 Whh,
// 700..827 Wih[:,400:528] (ctx), rest zero.
__global__ void pack_wcat_kernel(const float* __restrict__ wih,
                                 const float* __restrict__ whh,
                                 bhalf* __restrict__ dst) {
  int i = blockIdx.x * blockDim.x + threadIdx.x;
  if (i >= 1200 * 864) return;
  int n = i / 864, k = i - n * 864;
  float v = 0.f;
  if (k < 400)      v = wih[n * 528 + k];
  else if (k < 700) v = whh[n * 300 + (k - 400)];
  else if (k < 828) v = wih[n * 528 + 400 + (k - 700)];
  dst[i] = (bhalf)v;
}

__global__ void pack_wproj_kernel(const float* __restrict__ src, bhalf* __restrict__ dst) {
  int i = blockIdx.x * blockDim.x + threadIdx.x;
  if (i >= 48 * 448) return;
  int n = i / 448, k = i - n * 448;
  dst[i] = (n < 33 && k < 428) ? (bhalf)src[n * 428 + k] : (bhalf)0.f;
}

__global__ void add_bias_kernel(const float* a, const float* b, float* dst, int n) {
  int i = blockIdx.x * blockDim.x + threadIdx.x;
  if (i < n) dst[i] = a[i] + b[i];
}

// ---------------------------------------------------------------------------
// Phase B: persistent LSTM recurrence. grid={fwd,bwd}; 512 threads (16 waves).
// h (bf16, 32x320), c (f32) and gates (32x1200 f32) live in LDS.
// Whh (bf16, 1200x320) streamed from L2 as B-fragments every step.
// preG[t] row (76.8KB) is staged into LDS with CDNA5 async load-to-LDS,
// issued one step ahead and overlapped with the activation phase.
// ---------------------------------------------------------------------------
__global__ __launch_bounds__(512) void lstm_seq_kernel(
    const bhalf* __restrict__ preG_f, const bhalf* __restrict__ preG_b,
    const bhalf* __restrict__ whh_f, const bhalf* __restrict__ whh_b,
    bhalf* __restrict__ hseq, int T) {
  extern __shared__ char smem[];
  float* gates = (float*)smem;               // 32*1200 f32
  float* cst   = gates + 32 * 1200;          // 32*300  f32
  bhalf* hb    = (bhalf*)(cst + 32 * 300);   // 32*320  bf16 (K-padded)
  bhalf* pgLds = hb + 32 * 320;              // 32*1200 bf16 async-staged preG row

  const int dir = blockIdx.x;
  const bhalf* preG = dir ? preG_b : preG_f;
  const bhalf* whh  = dir ? whh_b : whh_f;
  const int colOff = dir * 300;

  for (int u = threadIdx.x; u < 32 * 320; u += blockDim.x) hb[u] = (bhalf)0.f;
  for (int u = threadIdx.x; u < 32 * 300; u += blockDim.x) cst[u] = 0.f;

#if USE_ASYNC_LDS
  // stage preG[t0] (4800 x b128 chunks spread across the block)
  {
    int t0 = dir ? (T - 1) : 0;
    const bhalf* src = preG + (size_t)t0 * 38400;
    for (int ch = threadIdx.x; ch < 4800; ch += blockDim.x)
      async_b128(src + ch * 8, pgLds + ch * 8);
  }
#endif
  __syncthreads();

  for (int s = 0; s < T; ++s) {
    int t = dir ? (T - 1 - s) : s;
#if USE_ASYNC_LDS
    __builtin_amdgcn_s_wait_asynccnt(0);
    __syncthreads();  // pgLds == preG[t] visible to all waves
    mm32(hb, 320, whh, 320, nullptr, 0, pgLds, 1200, gates, 1200, 75, 10);
    __syncthreads();
    if (s + 1 < T) {  // overlap next row's copy with the activation phase
      int tn = dir ? (T - 2 - s) : (s + 1);
      const bhalf* src = preG + (size_t)tn * 38400;
      for (int ch = threadIdx.x; ch < 4800; ch += blockDim.x)
        async_b128(src + ch * 8, pgLds + ch * 8);
    }
#else
    mm32(hb, 320, whh, 320, nullptr, 0, preG + (size_t)t * 32 * 1200, 1200,
         gates, 1200, 75, 10);
    __syncthreads();
#endif
    for (int u = threadIdx.x; u < 32 * 300; u += blockDim.x) {
      int b = u / 300, j = u - b * 300;
      const float* g = gates + b * 1200;
      float gi = sigmoidf_(g[j]);
      float gf = sigmoidf_(g[300 + j]);
      float gg = tanhf(g[600 + j]);
      float go = sigmoidf_(g[900 + j]);
      float cc = gf * cst[u] + gi * gg;
      float hh = go * tanhf(cc);
      cst[u] = cc;
      hb[b * 320 + j] = (bhalf)hh;
      hseq[((size_t)t * 32 + b) * 600 + colOff + j] = (bhalf)hh;
    }
    __syncthreads();
  }
}

// ---------------------------------------------------------------------------
// Decoder: single persistent workgroup, 500 greedy steps.
// xcat (bf16 [32,864]) = [emb(400) | h(300) | ctx(128) | 0pad] in LDS; the
// cell / phi / proj GEMMs all read it with pointer offsets (weights are
// zero-padded in K so overlap reads are harmless).
// ---------------------------------------------------------------------------
__global__ __launch_bounds__(512) void decoder_kernel(
    const bhalf* __restrict__ embB, const bhalf* __restrict__ wCat,
    const float* __restrict__ cellBias,
    const bhalf* __restrict__ wPhi, const float* __restrict__ phiBv,
    const bhalf* __restrict__ wProj, const float* __restrict__ projBv,
    const float* __restrict__ keysF, const float* __restrict__ valsF,
    const float* __restrict__ h0, const float* __restrict__ c0,
    float* __restrict__ out) {
  extern __shared__ char smem[];
  float* gates = (float*)smem;                 // 38400 f32 (reused arena)
  float* cst   = gates + 38400;                // 9600 f32
  bhalf* xcat  = (bhalf*)(cst + 9600);         // 32*864 bf16
  int*   chs   = (int*)(xcat + 32 * 864);      // 32
  float* smax  = (float*)(chs + 32);           // 32
  float* q      = gates;                       // 32*128
  float* logits = gates + 6144;                // 32*48
  float* e      = gates + 10240;               // 32*256

  const int tid = threadIdx.x;

  for (int u = tid; u < 32 * 864; u += blockDim.x) xcat[u] = (bhalf)0.f;
  __syncthreads();
  for (int u = tid; u < 32 * 300; u += blockDim.x) {
    int b = u / 300, j = u - b * 300;
    xcat[b * 864 + 400 + j] = (bhalf)h0[j];
    cst[u] = c0[j];
  }
  if (tid < 32) chs[tid] = 5;  // SOS
  __syncthreads();

  auto attend = [&]() {
    // q = h @ phiW^T + phib   (A = xcat cols 400.. ; phiW zero-padded K 300->320)
    mm32(xcat + 400, 864, wPhi, 320, phiBv, 128, nullptr, 0, q, 128, 8, 10);
    __syncthreads();
    // e[b,t] = q[b,:] . keys[t,b,:]
    for (int p = tid; p < 32 * 256; p += blockDim.x) {
      int b = p >> 8, t = p & 255;
      const float* qa = q + b * 128;
      const float* kr = keysF + ((size_t)t * 32 + b) * 128;
      float s = 0.f;
      for (int k = 0; k < 128; ++k) s += qa[k] * kr[k];
      e[b * 256 + t] = s;
    }
    __syncthreads();
    if (tid < 32) {
      float mx = -1e30f;
      for (int t = 0; t < 256; ++t) mx = fmaxf(mx, e[tid * 256 + t]);
      float s = 0.f;
      for (int t = 0; t < 256; ++t) {
        float ex = __expf(e[tid * 256 + t] - mx);
        e[tid * 256 + t] = ex;
        s += ex;
      }
      smax[tid] = 1.f / s;
    }
    __syncthreads();
    // ctx[b,d] = sum_t a * vals -> xcat ctx region (bf16)
    for (int p = tid; p < 32 * 128; p += blockDim.x) {
      int b = p >> 7, d = p & 127;
      float s = 0.f;
      for (int t = 0; t < 256; ++t)
        s += e[b * 256 + t] * valsF[((size_t)t * 32 + b) * 128 + d];
      xcat[b * 864 + 700 + d] = (bhalf)(s * smax[b]);
    }
    __syncthreads();
  };

  attend();  // ctx0 = attend(h0)

  for (int step = 0; step < 500; ++step) {
    // embedding of previous char -> xcat emb region
    for (int u = tid; u < 32 * 400; u += blockDim.x) {
      int b = u / 400, k = u - b * 400;
      xcat[b * 864 + k] = embB[chs[b] * 400 + k];
    }
    __syncthreads();
    // fused LSTM-cell gates: [emb|h|ctx] @ [Wih|Whh]^T + (bih+bhh)
    mm32(xcat, 864, wCat, 864, cellBias, 1200, nullptr, 0, gates, 1200, 75, 27);
    __syncthreads();
    for (int u = tid; u < 32 * 300; u += blockDim.x) {
      int b = u / 300, j = u - b * 300;
      const float* g = gates + b * 1200;
      float gi = sigmoidf_(g[j]);
      float gf = sigmoidf_(g[300 + j]);
      float gg = tanhf(g[600 + j]);
      float go = sigmoidf_(g[900 + j]);
      float cc = gf * cst[u] + gi * gg;
      float hh = go * tanhf(cc);
      cst[u] = cc;
      xcat[b * 864 + 400 + j] = (bhalf)hh;
    }
    __syncthreads();
    // logits = [h|ctx] @ projW^T + projb (N padded 33->48; argmax over 33)
    mm32(xcat + 400, 864, wProj, 448, projBv, 33, nullptr, 0, logits, 48, 3, 14);
    __syncthreads();
    if (tid < 32) {
      float best = logits[tid * 48];
      int bi = 0;
      for (int j = 1; j < 33; ++j) {
        float v = logits[tid * 48 + j];
        if (v > best) { best = v; bi = j; }
      }
      chs[tid] = bi;
      out[step * 32 + tid] = (float)bi;
    }
    attend();  // ctx = attend(h_new); also syncs before next iteration
  }
}

// ---------------------------------------------------------------------------
// Host orchestration
// ---------------------------------------------------------------------------
extern "C" void kernel_launch(void* const* d_in, const int* in_sizes, int n_in,
                              void* d_out, int out_size, void* d_ws, size_t ws_size,
                              hipStream_t stream) {
  (void)in_sizes; (void)n_in; (void)out_size; (void)ws_size;
  const float* X = (const float*)d_in[0];
  const float *Wih[4][2], *Whh[4][2], *bih[4][2], *bhh[4][2];
  {
    int idx = 1;
    for (int l = 0; l < 4; ++l)
      for (int d = 0; d < 2; ++d) {
        Wih[l][d] = (const float*)d_in[idx++];
        Whh[l][d] = (const float*)d_in[idx++];
        bih[l][d] = (const float*)d_in[idx++];
        bhh[l][d] = (const float*)d_in[idx++];
      }
  }
  const float* keyW = (const float*)d_in[33];
  const float* keyB = (const float*)d_in[34];
  const float* valW = (const float*)d_in[35];
  const float* valB = (const float*)d_in[36];
  const float* phiW = (const float*)d_in[37];
  const float* phiB = (const float*)d_in[38];
  const float* emb  = (const float*)d_in[39];
  const float* h0   = (const float*)d_in[40];
  const float* c0   = (const float*)d_in[41];
  const float* cWih = (const float*)d_in[42];
  const float* cWhh = (const float*)d_in[43];
  const float* cBih = (const float*)d_in[44];
  const float* cBhh = (const float*)d_in[45];
  const float* prW  = (const float*)d_in[46];
  const float* prB  = (const float*)d_in[47];

  char* base = (char*)d_ws;
  size_t cur = 0;
  auto alloc = [&](size_t bytes) -> void* {
    void* p = base + cur;
    cur = (cur + bytes + 255) & ~(size_t)255;
    return p;
  };

  const int KpIn[4] = {64, 1216, 1216, 1216};
  const int Kin[4]  = {40, 1200, 1200, 1200};
  const int Ms[4]   = {65536, 32768, 16384, 8192};
  const int Ts[4]   = {2048, 1024, 512, 256};

  bhalf *wWihB[4][2], *wWhhB[4][2];
  for (int l = 0; l < 4; ++l)
    for (int d = 0; d < 2; ++d) {
      wWihB[l][d] = (bhalf*)alloc((size_t)1200 * KpIn[l] * 2);
      wWhhB[l][d] = (bhalf*)alloc((size_t)1200 * 320 * 2);
    }
  bhalf* wKeyB = (bhalf*)alloc((size_t)128 * 608 * 2);
  bhalf* wValB = (bhalf*)alloc((size_t)128 * 608 * 2);
  bhalf* wPhiB = (bhalf*)alloc((size_t)128 * 320 * 2);
  bhalf* wCatB = (bhalf*)alloc((size_t)1200 * 864 * 2);
  bhalf* wPrjB = (bhalf*)alloc((size_t)48 * 448 * 2);
  bhalf* embB  = (bhalf*)alloc((size_t)33 * 400 * 2);
  float* cBias = (float*)alloc((size_t)1200 * 4);
  bhalf* A0    = (bhalf*)alloc((size_t)65536 * 64 * 2);
  bhalf* Apack = (bhalf*)alloc((size_t)32768 * 1216 * 2);
  bhalf* preG[2];
  preG[0] = (bhalf*)alloc((size_t)65536 * 1200 * 2);
  preG[1] = (bhalf*)alloc((size_t)65536 * 1200 * 2);
  bhalf* hseq[4];
  for (int l = 0; l < 4; ++l) hseq[l] = (bhalf*)alloc((size_t)Ts[l] * 32 * 600 * 2);
  float* keysF = (float*)alloc((size_t)8192 * 128 * 4);
  float* valsF = (float*)alloc((size_t)8192 * 128 * 4);

  auto cdiv = [](size_t a, size_t b) -> unsigned { return (unsigned)((a + b - 1) / b); };

  // ---- weight packing (f32 -> K-padded bf16) ----
  for (int l = 0; l < 4; ++l)
    for (int d = 0; d < 2; ++d) {
      size_t tot = (size_t)1200 * KpIn[l];
      pack_f32_kernel<<<cdiv(tot, 256), 256, 0, stream>>>(Wih[l][d], wWihB[l][d],
                                                          Kin[l], KpIn[l], tot);
      tot = (size_t)1200 * 320;
      pack_f32_kernel<<<cdiv(tot, 256), 256, 0, stream>>>(Whh[l][d], wWhhB[l][d],
                                                          300, 320, tot);
    }
  pack_f32_kernel<<<cdiv((size_t)128 * 608, 256), 256, 0, stream>>>(keyW, wKeyB, 600, 608, (size_t)128 * 608);
  pack_f32_kernel<<<cdiv((size_t)128 * 608, 256), 256, 0, stream>>>(valW, wValB, 600, 608, (size_t)128 * 608);
  pack_f32_kernel<<<cdiv((size_t)128 * 320, 256), 256, 0, stream>>>(phiW, wPhiB, 300, 320, (size_t)128 * 320);
  pack_f32_kernel<<<cdiv((size_t)33 * 400, 256), 256, 0, stream>>>(emb, embB, 400, 400, (size_t)33 * 400);
  pack_wcat_kernel<<<cdiv((size_t)1200 * 864, 256), 256, 0, stream>>>(cWih, cWhh, wCatB);
  pack_wproj_kernel<<<cdiv((size_t)48 * 448, 256), 256, 0, stream>>>(prW, wPrjB);
  add_bias_kernel<<<cdiv(1200, 256), 256, 0, stream>>>(cBih, cBhh, cBias, 1200);

  // ---- layer 0 input pack: X [2048,32,40] -> bf16 [65536, 64] ----
  pack_f32_kernel<<<cdiv((size_t)65536 * 64, 256), 256, 0, stream>>>(
      X, A0, 40, 64, (size_t)65536 * 64);

  // gates + c + h + async-staged preG row (always reserved; 289KB <= 320KB/WGP)
  const size_t smemL = (size_t)32 * 1200 * 4 + 32 * 300 * 4 + 32 * 320 * 2 +
                       (size_t)32 * 1200 * 2;

  // ---- 4 pyramidal BiLSTM layers ----
  for (int l = 0; l < 4; ++l) {
    const bhalf* Ain = (l == 0) ? A0 : Apack;
    if (l > 0) {
      size_t tot = (size_t)Ts[l] * 32 * 1216;
      pyramid_pack_kernel<<<cdiv(tot, 256), 256, 0, stream>>>(hseq[l - 1], Apack, Ts[l]);
    }
    for (int d = 0; d < 2; ++d) {
      dim3 g(cdiv(1200, 128), Ms[l] / 128);
      wmma_gemm_kernel<<<g, 256, 0, stream>>>(Ain, wWihB[l][d], bih[l][d], bhh[l][d],
                                              nullptr, preG[d], 1200, KpIn[l]);
    }
    lstm_seq_kernel<<<2, 512, smemL, stream>>>(preG[0], preG[1], wWhhB[l][0],
                                               wWhhB[l][1], hseq[l], Ts[l]);
  }

  // ---- key/value projections: [8192, 608] @ [128,608]^T ----
  {
    size_t tot = (size_t)8192 * 608;
    pack_bf16_kernel<<<cdiv(tot, 256), 256, 0, stream>>>(hseq[3], Apack, 600, 608, tot);
    dim3 g(1, 8192 / 128);
    wmma_gemm_kernel<<<g, 256, 0, stream>>>(Apack, wKeyB, keyB, nullptr, keysF, nullptr, 128, 608);
    wmma_gemm_kernel<<<g, 256, 0, stream>>>(Apack, wValB, valB, nullptr, valsF, nullptr, 128, 608);
  }

  // ---- greedy attention decoder ----
  const size_t smemD = (size_t)38400 * 4 + 9600 * 4 + 32 * 864 * 2 + 32 * 4 + 32 * 4;
  decoder_kernel<<<1, 512, smemD, stream>>>(embB, wCatB, cBias, wPhiB, phiB, wPrjB, prB,
                                            keysF, valsF, h0, c0, (float*)d_out);
}
